// GraphConvolution_16217796509779
// MI455X (gfx1250) — matrix operand order repaired and checked by reading, not compiled, not run on hardware
//
#include <hip/hip_runtime.h>

// ---------------------------------------------------------------------------
// RGCN layer for MI455X (gfx1250, wave32).
// Pipeline:
//   1. prep_B  : W_F = einsum('rb,bij->rij') combined + packed to f16 in the
//                WMMA B-operand lane layout (ISA 7.12.2).
//   2. gemm    : FW[s,n,:] = X[n,:] @ W_F[s]  via v_wmma_f32_16x16x32_f16,
//                f32 accumulate, stored as f16 (halves SpMM gather traffic).
//   3. hist/scan/scatter : build CSR-by-destination-row from COO (atomics only
//                on 400KB of counters, not on the 25MB output).
//   4. rowsum  : one wave32 per output row; per edge one coalesced 128B load
//                of an FW row, f32 accumulate, bias + ReLU on store.
// ---------------------------------------------------------------------------

typedef __attribute__((ext_vector_type(16))) _Float16 v16h;
typedef __attribute__((ext_vector_type(8)))  float    v8f;

#define F_DIM 128
#define O_DIM 64
#define S_REL 8
#define B_BAS 4

// ---------------------------------------------------------------------------
// Kernel 1: combine bases -> per-relation weights, pack into WMMA B layout.
// Bw layout: [s][coltile t=0..3][kchunk kc=0..3][lane=0..31][i=0..15] f16
//   For B (32x16, 16-bit): lanes 0-15 hold K=0..7,16..23, lanes 16-31 hold K=8..15,24..31,
//   column = lane&15, elems 0..7 / 8..15 walk the two K runs. (ISA 7.12.4 pattern)
// ---------------------------------------------------------------------------
__global__ void rgcn_prep_B(const float* __restrict__ bases,  // [B,F,O]
                            const float* __restrict__ comp,   // [S,B]
                            _Float16* __restrict__ Bw)
{
    int tid = blockIdx.x * blockDim.x + threadIdx.x;
    if (tid >= S_REL * 4 * 4 * 32 * 16) return;
    int i    = tid & 15;
    int lane = (tid >> 4) & 31;
    int kc   = (tid >> 9) & 3;
    int t    = (tid >> 11) & 3;
    int s    = tid >> 13;
    int khalf = lane >> 4;
    int col   = t * 16 + (lane & 15);
    int k     = kc * 32 + khalf * 8 + (i & 7) + (i >> 3) * 16;      // B layout: elems 0..7 = K 8h..8h+7, elems 8..15 = K 16+8h..23+8h
    float acc = 0.f;
#pragma unroll
    for (int b = 0; b < B_BAS; ++b)
        acc += comp[s * B_BAS + b] * bases[((size_t)b * F_DIM + k) * O_DIM + col];
    Bw[tid] = (_Float16)acc;
}

// ---------------------------------------------------------------------------
// Kernel 2: FW[s*N+n, o] = sum_k X[n,k] * W_F[s,k,o], WMMA f16->f32.
// Block = 256 threads = 8 waves; wave w handles relation s=w for one 16-node
// tile, computing a 16x64 strip: 4 coltiles x 4 kchunks = 16 WMMAs.
//
// A loads are UNGUARDED on purpose: the row pointer is clamped for
// out-of-range rows, and since A-row m only contributes to D-row m, garbage
// rows produce garbage accumulator rows that the guarded store never writes.
// This keeps the load path as coalesced b128s with no exec-mask churn.
// ---------------------------------------------------------------------------
__global__ void __launch_bounds__(256)
rgcn_gemm(const float* __restrict__ X,        // [N, F]
          const _Float16* __restrict__ Bw,    // packed B operands
          _Float16* __restrict__ FW,          // [S*N, O] f16
          int N)
{
    const int lane  = threadIdx.x & 31;
    const int s     = threadIdx.x >> 5;       // relation = wave id (S_REL == 8)
    const int tile  = blockIdx.x;
    const int row   = tile * 16 + (lane & 15);
    const int khalf = lane >> 4;
    const bool rok  = (row < N);
    const float* xrow = X + (size_t)(rok ? row : 0) * F_DIM;

    v8f c[4] = {{}, {}, {}, {}};

#pragma unroll
    for (int kc = 0; kc < 4; ++kc) {
        // A fragment, 16-bit A 16x32 layout (ISA 7.12.2):
        //   lane<16:  elems 0..7 = K 0..7,  elems 8..15 = K 16..23
        //   lane>=16: elems 0..7 = K 8..15, elems 8..15 = K 24..31
        const int kA = kc * 32 + khalf * 8;            // multiple of 8 floats -> 32B aligned
        const float4* xp0 = (const float4*)(xrow + kA);
        const float4* xp1 = (const float4*)(xrow + kA + 16);
        float4 p0 = xp0[0];
        float4 p1 = xp0[1];
        float4 p2 = xp1[0];
        float4 p3 = xp1[1];
        v16h a;
        a[0]  = (_Float16)p0.x;  a[1]  = (_Float16)p0.y;
        a[2]  = (_Float16)p0.z;  a[3]  = (_Float16)p0.w;
        a[4]  = (_Float16)p1.x;  a[5]  = (_Float16)p1.y;
        a[6]  = (_Float16)p1.z;  a[7]  = (_Float16)p1.w;
        a[8]  = (_Float16)p2.x;  a[9]  = (_Float16)p2.y;
        a[10] = (_Float16)p2.z;  a[11] = (_Float16)p2.w;
        a[12] = (_Float16)p3.x;  a[13] = (_Float16)p3.y;
        a[14] = (_Float16)p3.z;  a[15] = (_Float16)p3.w;

        // B fragments: 32B contiguous per lane, coltile stride = 2048 halves.
        const v16h* bp = (const v16h*)(Bw +
            ((((size_t)s * 4 + 0) * 4 + kc) * 32 + lane) * 16);
        v16h b0 = bp[0];
        v16h b1 = bp[128];
        v16h b2 = bp[256];
        v16h b3 = bp[384];
        c[0] = __builtin_amdgcn_wmma_f32_16x16x32_f16(false, a, false, b0, (short)0, c[0], false, false);
        c[1] = __builtin_amdgcn_wmma_f32_16x16x32_f16(false, a, false, b1, (short)0, c[1], false, false);
        c[2] = __builtin_amdgcn_wmma_f32_16x16x32_f16(false, a, false, b2, (short)0, c[2], false, false);
        c[3] = __builtin_amdgcn_wmma_f32_16x16x32_f16(false, a, false, b3, (short)0, c[3], false, false);
    }

    // C/D 16x16 f32 layout: VGPR i -> (M=i, N=lane) lanes 0-15; (M=i+8, N=lane-16).
    const int n_in_tile = lane & 15;
    const int mbase = (lane < 16) ? 0 : 8;
#pragma unroll
    for (int t = 0; t < 4; ++t) {
#pragma unroll
        for (int i = 0; i < 8; ++i) {
            int r = tile * 16 + mbase + i;
            if (r < N)
                FW[((size_t)s * N + r) * O_DIM + t * 16 + n_in_tile] = (_Float16)c[t][i];
        }
    }
}

// ---------------------------------------------------------------------------
// CSR build: histogram -> exclusive scan -> scatter.
// ---------------------------------------------------------------------------
__global__ void rgcn_zero(int* __restrict__ p, int n)
{
    int i = blockIdx.x * blockDim.x + threadIdx.x;
    if (i < n) p[i] = 0;
}

__global__ void rgcn_hist(const int* __restrict__ rows, int* __restrict__ hist, int E)
{
    int e = blockIdx.x * blockDim.x + threadIdx.x;
    if (e < E) atomicAdd(&hist[rows[e]], 1);
}

__global__ void rgcn_scan(const int* __restrict__ hist,
                          int* __restrict__ offs,
                          int* __restrict__ cursor, int N)
{
    __shared__ int sm[1024];
    __shared__ int running;
    const int t = threadIdx.x;
    if (t == 0) running = 0;
    __syncthreads();
    for (int base = 0; base < N; base += 1024) {
        int v = (base + t < N) ? hist[base + t] : 0;
        sm[t] = v;
        __syncthreads();
        for (int off = 1; off < 1024; off <<= 1) {   // Hillis-Steele inclusive
            int x = (t >= off) ? sm[t - off] : 0;
            __syncthreads();
            sm[t] += x;
            __syncthreads();
        }
        int incl = sm[t];
        int r = running;
        if (base + t < N) {
            int excl = r + incl - v;
            offs[base + t]   = excl;
            cursor[base + t] = excl;
        }
        __syncthreads();
        if (t == 1023) running = r + incl;
        __syncthreads();
    }
    if (t == 0) offs[N] = running;
}

__global__ void rgcn_scatter(const int* __restrict__ rows,
                             const int* __restrict__ cols,
                             const float* __restrict__ vals,
                             int* __restrict__ cursor,
                             int* __restrict__ sCol,
                             float* __restrict__ sVal, int E)
{
    int e = blockIdx.x * blockDim.x + threadIdx.x;
    if (e < E) {
        int p = atomicAdd(&cursor[rows[e]], 1);
        sCol[p] = cols[e];
        sVal[p] = vals[e];
    }
}

// ---------------------------------------------------------------------------
// Kernel 4: one wave32 per output row; lane owns output cols {2L, 2L+1}.
// Per edge: one coalesced 128B wave load of an FW row (f16), f32 accumulate.
// ---------------------------------------------------------------------------
__global__ void __launch_bounds__(256)
rgcn_rowsum(const int* __restrict__ offs,
            const int* __restrict__ sCol,
            const float* __restrict__ sVal,
            const _Float16* __restrict__ FW,
            const float* __restrict__ bias,
            float* __restrict__ out, int N)
{
    int gtid = blockIdx.x * blockDim.x + threadIdx.x;
    int row  = gtid >> 5;
    int lane = gtid & 31;
    if (row >= N) return;

    int start = offs[row];
    int end   = offs[row + 1];
    float a0 = 0.f, a1 = 0.f;
    const unsigned int* fw32 = (const unsigned int*)FW;   // 32 dwords per FW row
    for (int j = start; j < end; ++j) {
        int   col = sCol[j];
        float val = sVal[j];
        union { unsigned int u; _Float16 h[2]; } cv;
        cv.u = fw32[(size_t)col * 32 + lane];
        a0 += val * (float)cv.h[0];
        a1 += val * (float)cv.h[1];
    }
    float r0 = a0 + bias[2 * lane];
    float r1 = a1 + bias[2 * lane + 1];
    out[(size_t)row * O_DIM + 2 * lane]     = r0 > 0.f ? r0 : 0.f;
    out[(size_t)row * O_DIM + 2 * lane + 1] = r1 > 0.f ? r1 : 0.f;
}

// ---------------------------------------------------------------------------
extern "C" void kernel_launch(void* const* d_in, const int* in_sizes, int n_in,
                              void* d_out, int out_size, void* d_ws, size_t ws_size,
                              hipStream_t stream)
{
    const float* X      = (const float*)d_in[0];
    const int*   A_rows = (const int*)  d_in[1];
    const int*   A_cols = (const int*)  d_in[2];
    const float* A_vals = (const float*)d_in[3];
    const float* bases  = (const float*)d_in[4];
    const float* comp   = (const float*)d_in[5];
    const float* bias   = (const float*)d_in[6];
    float*       outf   = (float*)d_out;

    const int N = in_sizes[0] / F_DIM;
    const int E = in_sizes[1];

    // Workspace carve (256B aligned): FW f16 | Bw f16 | hist | offs | cursor | sCol | sVal
    char* ws = (char*)d_ws;
    size_t off = 0;
    auto carve = [&](size_t bytes) -> char* {
        char* p = ws + off;
        off = (off + bytes + 255) & ~(size_t)255;
        return p;
    };
    _Float16* FW     = (_Float16*)carve((size_t)S_REL * N * O_DIM * sizeof(_Float16));
    _Float16* Bw     = (_Float16*)carve((size_t)S_REL * F_DIM * O_DIM * sizeof(_Float16));
    int*      hist   = (int*)     carve((size_t)N * sizeof(int));
    int*      offs   = (int*)     carve((size_t)(N + 1) * sizeof(int));
    int*      cursor = (int*)     carve((size_t)N * sizeof(int));
    int*      sCol   = (int*)     carve((size_t)E * sizeof(int));
    float*    sVal   = (float*)   carve((size_t)E * sizeof(float));
    (void)ws_size; (void)n_in; (void)out_size;

    rgcn_zero<<<(N + 255) / 256, 256, 0, stream>>>(hist, N);

    const int bwTotal = S_REL * 4 * 4 * 32 * 16;
    rgcn_prep_B<<<(bwTotal + 255) / 256, 256, 0, stream>>>(bases, comp, Bw);

    rgcn_gemm<<<(N + 15) / 16, 256, 0, stream>>>(X, Bw, FW, N);

    rgcn_hist<<<(E + 255) / 256, 256, 0, stream>>>(A_rows, hist, E);
    rgcn_scan<<<1, 1024, 0, stream>>>(hist, offs, cursor, N);
    rgcn_scatter<<<(E + 255) / 256, 256, 0, stream>>>(A_rows, A_cols, A_vals,
                                                      cursor, sCol, sVal, E);

    long long waves  = N;
    long long blocks = (waves * 32 + 255) / 256;
    rgcn_rowsum<<<(int)blocks, 256, 0, stream>>>(offs, sCol, sVal, FW, bias, outf, N);
}
